// TextLSTM_19937238188552
// MI455X (gfx1250) — matrix operand order repaired and checked by reading, not compiled, not run on hardware
//
#include <hip/hip_runtime.h>
#include <stdint.h>

#define T_DIM 512
#define B_DIM 256
#define D_DIM 512
#define H_DIM 512
#define TB_DIM (T_DIM * B_DIM)   // 131072
#define N4H (4 * H_DIM)          // 2048

typedef __attribute__((ext_vector_type(16))) __bf16 v16bf;
typedef __attribute__((ext_vector_type(8)))  float  v8f;

// matches the builtin's parameter type: 16-byte int vector
typedef int v4i_vs __attribute__((vector_size(16)));
typedef __attribute__((address_space(1))) v4i_vs* as1_v4i;
typedef __attribute__((address_space(3))) v4i_vs* as3_v4i;

union Frag {
    v16bf v;
    uint4 q[2];
};

__device__ __forceinline__ unsigned short f2bf(float f) {
    union { float f; unsigned int u; } x;
    x.f = f;
    unsigned int u = x.u;
    u += 0x7fffu + ((u >> 16) & 1u);   // round-to-nearest-even
    return (unsigned short)(u >> 16);
}

// ---------------------------------------------------------------------------
// Cooperative tile staging: global -> LDS, n16 chunks of 16 bytes.
// Prefers gfx1250 GLOBAL_LOAD_ASYNC_TO_LDS_B128 (ASYNCcnt path); falls back
// to register-relayed b128 copies if the builtins are unavailable.
// Caller must __syncthreads() afterwards. blockDim.x == 256 assumed.
// ---------------------------------------------------------------------------
#if __has_builtin(__builtin_amdgcn_global_load_async_to_lds_b128) && \
    __has_builtin(__builtin_amdgcn_s_wait_asynccnt)
#define HAVE_ASYNC_LDS 1
#else
#define HAVE_ASYNC_LDS 0
#endif

__device__ __forceinline__ void tile_to_lds(unsigned short* dst,
                                            const unsigned short* src,
                                            int n16) {
#if HAVE_ASYNC_LDS
    // low 32 bits of a generic LDS pointer are the LDS byte offset
    unsigned long long gbase = (unsigned long long)(uintptr_t)src;
    unsigned int       lbase = (unsigned int)(uintptr_t)dst;
    for (int i = threadIdx.x; i < n16; i += 256) {
        __builtin_amdgcn_global_load_async_to_lds_b128(
            (as1_v4i)(uintptr_t)(gbase + (unsigned long long)i * 16),
            (as3_v4i)(uintptr_t)(lbase + (unsigned int)i * 16),
            0, 0);
    }
    __builtin_amdgcn_s_wait_asynccnt(0);
#else
    const uint4* s = (const uint4*)src;
    uint4*       d = (uint4*)dst;
    for (int i = threadIdx.x; i < n16; i += 256)
        d[i] = s[i];
#endif
}

// ---------------------------------------------------------------------------
// fp32 -> bf16 conversion (vectorized x4)
// ---------------------------------------------------------------------------
__global__ __launch_bounds__(256)
void k_cvt_bf16(const float* __restrict__ src, unsigned short* __restrict__ dst, int n) {
    int i = (blockIdx.x * blockDim.x + threadIdx.x) * 4;
    if (i < n) {
        float4 v = *(const float4*)(src + i);
        unsigned int lo = (unsigned int)f2bf(v.x) | ((unsigned int)f2bf(v.y) << 16);
        unsigned int hi = (unsigned int)f2bf(v.z) | ((unsigned int)f2bf(v.w) << 16);
        *(uint2*)(dst + i) = make_uint2(lo, hi);
    }
}

// zero h double-buffers (bf16) and cell state (fp32)
__global__ __launch_bounds__(256)
void k_init_state(unsigned short* hb0, unsigned short* hb1, float* c) {
    int i = blockIdx.x * blockDim.x + threadIdx.x;   // B*H threads
    hb0[i] = 0;
    hb1[i] = 0;
    c[i]   = 0.0f;
}

// ---------------------------------------------------------------------------
// Phase 1: P[t][g][b][h] = X[t,b,:] . W_g[h,:]  (+ input bias for f,i,o)
// One big GEMM: M = T*B, N = 4H, K = D.  Block = 256 thr = 8 waves.
// Block tile: 32 M-rows (LDS) x 256 N-cols.  Each wave: 2x2 register-blocked
// 16x16 tiles -> 4 wmma per 8 b128 loads (2x the wmma:vmem ratio of 1x1).
// ---------------------------------------------------------------------------
__global__ __launch_bounds__(256)
void k_gemm_x(const unsigned short* __restrict__ Xb,     // [TB, D] bf16
              const unsigned short* __restrict__ Wcat,   // [4H, D] bf16
              const float* __restrict__ bf,
              const float* __restrict__ bi,
              const float* __restrict__ bo,
              float* __restrict__ P) {                   // [T][4][B][H] f32
    __shared__ unsigned short sA[32 * D_DIM];            // 32 KB

    const int m0 = blockIdx.x * 32;
    tile_to_lds(sA, Xb + (size_t)m0 * D_DIM, (32 * D_DIM) / 8);
    __syncthreads();

    const int lane = threadIdx.x & 31;
    const int wave = threadIdx.x >> 5;
    const int half = lane >> 4;
    const int l16  = lane & 15;
    const int n0   = blockIdx.y * 256 + wave * 32;

    v8f acc[2][2] = {};
    const unsigned short* wptr0 = Wcat + (size_t)(n0 + l16) * D_DIM + half * 8;
    const unsigned short* wptr1 = Wcat + (size_t)(n0 + 16 + l16) * D_DIM + half * 8;
    const unsigned short* aptr0 = sA + l16 * D_DIM + half * 8;
    const unsigned short* aptr1 = sA + (16 + l16) * D_DIM + half * 8;

    for (int k0 = 0; k0 < D_DIM; k0 += 32) {
        Frag a0, a1, b0, b1;
        a0.q[0] = *(const uint4*)(aptr0 + k0);
        a0.q[1] = *(const uint4*)(aptr0 + k0 + 16);
        a1.q[0] = *(const uint4*)(aptr1 + k0);
        a1.q[1] = *(const uint4*)(aptr1 + k0 + 16);
        b0.q[0] = *(const uint4*)(wptr0 + k0);
        b0.q[1] = *(const uint4*)(wptr0 + k0 + 16);
        b1.q[0] = *(const uint4*)(wptr1 + k0);
        b1.q[1] = *(const uint4*)(wptr1 + k0 + 16);
        acc[0][0] = __builtin_amdgcn_wmma_f32_16x16x32_bf16(false, a0.v, false, b0.v, (short)0, acc[0][0], false, false);
        acc[0][1] = __builtin_amdgcn_wmma_f32_16x16x32_bf16(false, a0.v, false, b1.v, (short)0, acc[0][1], false, false);
        acc[1][0] = __builtin_amdgcn_wmma_f32_16x16x32_bf16(false, a1.v, false, b0.v, (short)0, acc[1][0], false, false);
        acc[1][1] = __builtin_amdgcn_wmma_f32_16x16x32_bf16(false, a1.v, false, b1.v, (short)0, acc[1][1], false, false);
    }

#pragma unroll
    for (int ni = 0; ni < 2; ++ni) {
        const int col = n0 + ni * 16 + l16;   // output column in [4H]
        const int g = col >> 9;               // gate 0..3
        const int h = col & 511;
        float bias = 0.0f;
        if (g == 0) bias = bf[h];
        else if (g == 1) bias = bi[h];
        else if (g == 2) bias = bo[h];
#pragma unroll
        for (int mi = 0; mi < 2; ++mi) {
#pragma unroll
            for (int r = 0; r < 8; ++r) {
                int m  = m0 + mi * 16 + r + 8 * half;  // row index in [TB]
                int t  = m >> 8;                       // B = 256
                int bb = m & 255;
                P[(((size_t)t * 4 + g) * B_DIM + bb) * H_DIM + h] = acc[mi][ni][r] + bias;
            }
        }
    }
}

// ---------------------------------------------------------------------------
// Phase 2: one timestep.  a = h_prev @ W_aa^T + b_a  (bf16 WMMA, fp32 acc),
// fused with the gate nonlinearity / cell update.  h double-buffered in bf16.
// Grid: (B/16, H/128), 256 threads = 8 waves, one 16x16 tile per wave.
// ---------------------------------------------------------------------------
__global__ __launch_bounds__(256)
void k_step(const unsigned short* __restrict__ Hprev,    // [B, H] bf16
            unsigned short* __restrict__ Hnext,          // [B, H] bf16
            const unsigned short* __restrict__ Waa,      // [H, H] bf16
            const float* __restrict__ ba,
            const float* __restrict__ P,                 // [T][4][B][H] f32
            float* __restrict__ Cst,                     // [B, H] f32
            float* __restrict__ out,                     // d_out
            int t) {
    __shared__ unsigned short sA[16 * H_DIM];            // 16 KB

    const int m0 = blockIdx.x * 16;                      // batch rows
    tile_to_lds(sA, Hprev + (size_t)m0 * H_DIM, (16 * H_DIM) / 8);
    __syncthreads();

    const int lane = threadIdx.x & 31;
    const int wave = threadIdx.x >> 5;
    const int half = lane >> 4;
    const int l16  = lane & 15;
    const int n0   = blockIdx.y * 128 + wave * 16;
    const int h    = n0 + l16;

    v8f acc = {};
    const unsigned short* wptr = Waa + (size_t)h * H_DIM + half * 8;
    const unsigned short* aptr = sA + l16 * H_DIM + half * 8;

    for (int k0 = 0; k0 < H_DIM; k0 += 32) {
        Frag a, b;
        a.q[0] = *(const uint4*)(aptr + k0);
        a.q[1] = *(const uint4*)(aptr + k0 + 16);
        b.q[0] = *(const uint4*)(wptr + k0);
        b.q[1] = *(const uint4*)(wptr + k0 + 16);
        acc = __builtin_amdgcn_wmma_f32_16x16x32_bf16(
            false, a.v, false, b.v, (short)0, acc, false, false);
    }

    const float biasa = ba[h];
    const float* Pf = P + ((size_t)t * 4 + 0) * B_DIM * H_DIM;
    const float* Pi = P + ((size_t)t * 4 + 1) * B_DIM * H_DIM;
    const float* Po = P + ((size_t)t * 4 + 2) * B_DIM * H_DIM;
    const float* Pa = P + ((size_t)t * 4 + 3) * B_DIM * H_DIM;
    float* outt = out + (size_t)t * B_DIM * H_DIM;

#pragma unroll
    for (int r = 0; r < 8; ++r) {
        int brow = m0 + r + 8 * half;
        size_t idx = (size_t)brow * H_DIM + h;
        float a  = acc[r] + biasa;                 // shared recurrent term
        float xf = Pf[idx] + a;
        float xi = Pi[idx] + a;
        float xo = Po[idx] + a;
        float xg = Pa[idx] + a;
        float ft = 1.0f / (1.0f + __expf(-xf));
        float it = 1.0f / (1.0f + __expf(-xi));
        float ot = 1.0f / (1.0f + __expf(-xo));
        float gt = tanhf(xg);
        float cn = ft * Cst[idx] + it * gt;
        float hn = ot * tanhf(cn);
        Cst[idx]  = cn;
        outt[idx] = hn;
        Hnext[idx] = f2bf(hn);
        if (t == T_DIM - 1)
            out[(size_t)T_DIM * B_DIM * H_DIM + idx] = hn;   // h_n
    }
}

// ---------------------------------------------------------------------------
extern "C" void kernel_launch(void* const* d_in, const int* in_sizes, int n_in,
                              void* d_out, int out_size, void* d_ws, size_t ws_size,
                              hipStream_t stream) {
    (void)in_sizes; (void)n_in; (void)out_size; (void)ws_size;

    const float* X   = (const float*)d_in[0];
    const float* Wfx = (const float*)d_in[1];
    const float* Wix = (const float*)d_in[2];
    const float* Wox = (const float*)d_in[3];
    const float* Wax = (const float*)d_in[4];
    const float* Waa = (const float*)d_in[5];
    const float* bf  = (const float*)d_in[6];
    const float* bi  = (const float*)d_in[7];
    const float* bo  = (const float*)d_in[8];
    const float* ba  = (const float*)d_in[9];
    float* out = (float*)d_out;

    // workspace carve-up (all regions naturally 256B-aligned)
    char* ws = (char*)d_ws;
    size_t off = 0;
    unsigned short* Xb   = (unsigned short*)(ws + off); off += (size_t)TB_DIM * D_DIM * 2;  // 134 MB
    unsigned short* Wcat = (unsigned short*)(ws + off); off += (size_t)N4H * D_DIM * 2;     // 2 MB
    unsigned short* Waab = (unsigned short*)(ws + off); off += (size_t)H_DIM * H_DIM * 2;   // 0.5 MB
    unsigned short* Hb0  = (unsigned short*)(ws + off); off += (size_t)B_DIM * H_DIM * 2;
    unsigned short* Hb1  = (unsigned short*)(ws + off); off += (size_t)B_DIM * H_DIM * 2;
    float*          Cst  = (float*)(ws + off);          off += (size_t)B_DIM * H_DIM * 4;
    float*          P    = (float*)(ws + off);          off += (size_t)4 * T_DIM * B_DIM * H_DIM * 4; // 1.07 GB

    // --- conversions to bf16 ---
    {
        int n = TB_DIM * D_DIM;
        k_cvt_bf16<<<n / 1024, 256, 0, stream>>>(X, Xb, n);
    }
    {
        int n = H_DIM * D_DIM;
        k_cvt_bf16<<<n / 1024, 256, 0, stream>>>(Wfx, Wcat + 0 * (size_t)H_DIM * D_DIM, n);
        k_cvt_bf16<<<n / 1024, 256, 0, stream>>>(Wix, Wcat + 1 * (size_t)H_DIM * D_DIM, n);
        k_cvt_bf16<<<n / 1024, 256, 0, stream>>>(Wox, Wcat + 2 * (size_t)H_DIM * D_DIM, n);
        k_cvt_bf16<<<n / 1024, 256, 0, stream>>>(Wax, Wcat + 3 * (size_t)H_DIM * D_DIM, n);
        k_cvt_bf16<<<(H_DIM * H_DIM) / 1024, 256, 0, stream>>>(Waa, Waab, H_DIM * H_DIM);
    }
    k_init_state<<<(B_DIM * H_DIM) / 256, 256, 0, stream>>>(Hb0, Hb1, Cst);

    // --- phase 1: big input-projection GEMM ---
    {
        dim3 grid(TB_DIM / 32, N4H / 256);   // (4096, 8)
        k_gemm_x<<<grid, 256, 0, stream>>>(Xb, Wcat, bf, bi, bo, P);
    }

    // --- phase 2: sequential recurrence, one fused kernel per step ---
    {
        dim3 grid(B_DIM / 16, H_DIM / 128);  // (16, 4)
        for (int t = 0; t < T_DIM; ++t) {
            const unsigned short* hp = (t & 1) ? Hb1 : Hb0;
            unsigned short*       hn = (t & 1) ? Hb0 : Hb1;
            k_step<<<grid, 256, 0, stream>>>(hp, hn, Waab, ba, P, Cst, out, t);
        }
    }
}
